// EncoderLayer_foridx_47390669144536
// MI455X (gfx1250) — matrix operand order repaired and checked by reading, not compiled, not run on hardware
//
#include <hip/hip_runtime.h>
#include <math.h>

#define H_DIM 1024
#define F_DIM 4096
#define B_SZ  8
#define N_SEQ 2048
#define T_TOK (B_SZ * N_SEQ)

typedef __attribute__((ext_vector_type(16))) __bf16 v16bf;
typedef __attribute__((ext_vector_type(8)))  float  v8f;

#if __has_builtin(__builtin_amdgcn_global_load_async_to_lds_b128) && \
    __has_builtin(__builtin_amdgcn_s_wait_asynccnt)
#define USE_ASYNC_LDS 1
typedef int v4i __attribute__((vector_size(16)));            // "V4i" builtin vector type
#define GPTR(p) ((__attribute__((address_space(1))) v4i*)(p))  // global int4*
#define LPTR(p) ((__attribute__((address_space(3))) v4i*)(p))  // LDS int4*
#else
#define USE_ASYNC_LDS 0
#endif

__device__ __forceinline__ unsigned short f2bf(float f) {
  unsigned int u = __float_as_uint(f);
  u += 0x7fffu + ((u >> 16) & 1u);        // round-to-nearest-even
  return (unsigned short)(u >> 16);
}
__device__ __forceinline__ float bf2f(unsigned short h) {
  return __uint_as_float(((unsigned int)h) << 16);
}

// ---------------------------------------------------------------------------
// Weight transpose + f32->bf16 convert:  W[K][N] (row major)  ->  Wt[N][K]
// ---------------------------------------------------------------------------
__global__ __launch_bounds__(256)
void transpose_cvt_bf16(const float* __restrict__ W, unsigned short* __restrict__ Wt,
                        int K, int N) {
  __shared__ float tile[32][33];
  const int k0 = blockIdx.y * 32;
  const int n0 = blockIdx.x * 32;
  for (int i = threadIdx.y; i < 32; i += 8)
    tile[i][threadIdx.x] = W[(size_t)(k0 + i) * N + (n0 + threadIdx.x)];
  __syncthreads();
  for (int i = threadIdx.y; i < 32; i += 8)
    Wt[(size_t)(n0 + i) * K + (k0 + threadIdx.x)] = f2bf(tile[threadIdx.x][i]);
}

// ---------------------------------------------------------------------------
// LayerNorm over last dim (H_DIM=1024), one row per 256-thread block, bf16 out
// ---------------------------------------------------------------------------
__global__ __launch_bounds__(256)
void layernorm_bf16(const float* __restrict__ x, const float* __restrict__ g,
                    const float* __restrict__ b, unsigned short* __restrict__ out) {
  __shared__ float sh[256], sh2[256];
  const size_t row = blockIdx.x;
  const float* xr = x + row * H_DIM;
  float v[4], s = 0.f, s2 = 0.f;
#pragma unroll
  for (int i = 0; i < 4; ++i) {
    v[i] = xr[threadIdx.x + i * 256];
    s += v[i]; s2 += v[i] * v[i];
  }
  sh[threadIdx.x] = s; sh2[threadIdx.x] = s2;
  __syncthreads();
  for (int o = 128; o > 0; o >>= 1) {
    if (threadIdx.x < o) { sh[threadIdx.x] += sh[threadIdx.x + o]; sh2[threadIdx.x] += sh2[threadIdx.x + o]; }
    __syncthreads();
  }
  const float mean = sh[0] * (1.0f / H_DIM);
  const float var  = sh2[0] * (1.0f / H_DIM) - mean * mean;
  const float rstd = rsqrtf(var + 1e-5f);
  unsigned short* orow = out + row * H_DIM;
#pragma unroll
  for (int i = 0; i < 4; ++i) {
    const int c = threadIdx.x + i * 256;
    orow[c] = f2bf((v[i] - mean) * rstd * g[c] + b[c]);
  }
}

// ---------------------------------------------------------------------------
// In-place softmax over rows of length N_SEQ (bf16)
// ---------------------------------------------------------------------------
__global__ __launch_bounds__(256)
void softmax_inplace_bf16(unsigned short* __restrict__ S) {
  __shared__ float sh[256];
  unsigned short* r = S + (size_t)blockIdx.x * N_SEQ;
  float v[8], m = -3.0e38f;
#pragma unroll
  for (int i = 0; i < 8; ++i) { v[i] = bf2f(r[threadIdx.x + i * 256]); m = fmaxf(m, v[i]); }
  sh[threadIdx.x] = m; __syncthreads();
  for (int o = 128; o > 0; o >>= 1) {
    if (threadIdx.x < o) sh[threadIdx.x] = fmaxf(sh[threadIdx.x], sh[threadIdx.x + o]);
    __syncthreads();
  }
  m = sh[0]; __syncthreads();
  float s = 0.f;
#pragma unroll
  for (int i = 0; i < 8; ++i) { v[i] = expf(v[i] - m); s += v[i]; }
  sh[threadIdx.x] = s; __syncthreads();
  for (int o = 128; o > 0; o >>= 1) {
    if (threadIdx.x < o) sh[threadIdx.x] += sh[threadIdx.x + o];
    __syncthreads();
  }
  const float inv = 1.0f / sh[0];
#pragma unroll
  for (int i = 0; i < 8; ++i) r[threadIdx.x + i * 256] = f2bf(v[i] * inv);
}

// ---------------------------------------------------------------------------
// WMMA bf16 GEMM:  C[M][Nout] = A[M][K] * Bt[N][K]^T  (+ epilogue)
//   EPI 0: bf16 store            EPI 1: bf16 store transposed per batch (vT)
//   EPI 2: *scale -> bf16        EPI 3: +bias, GELU -> bf16
//   EPI 4: +bias +resid -> f32   EPI 5: +resid -> f32
// block tile 128x128, 8 waves (2Mx4N), wave tile 64x32, BK=32
// Double-buffered LDS; global->LDS via CDNA5 async DMA when available.
// ---------------------------------------------------------------------------
struct GemmParams {
  const unsigned short* A;
  const unsigned short* Bt;
  unsigned short* outBf;
  float* outF;
  const float* bias;
  const float* resid;
  long long sA, sB, sO, sR;   // per-batch element strides (blockIdx.z)
  int K;
  int Nout;
  float scale;
};

template<int EPI>
__global__ __launch_bounds__(256)
void gemm_bf16_wmma(GemmParams p) {
  constexpr int BM = 128, BN = 128, BK = 32;
  constexpr int LDT = BK + 8;                       // 40 elems = 80B (16B-aligned rows)
  __shared__ __align__(16) unsigned short sA[2][BM * LDT];
  __shared__ __align__(16) unsigned short sB[2][BN * LDT];

  const int tid  = threadIdx.x;
  const int lane = tid & 31;
  const int wave = tid >> 5;
  const int wm   = wave & 1;                        // wave row  (0..1)
  const int wn   = wave >> 1;                       // wave col  (0..3)
  const int bz   = blockIdx.z;

  const unsigned short* Ag  = p.A  + (size_t)bz * p.sA + (size_t)blockIdx.y * BM * p.K;
  const unsigned short* Btg = p.Bt + (size_t)bz * p.sB + (size_t)blockIdx.x * BN * p.K;

  v8f acc[4][2];
#pragma unroll
  for (int mt = 0; mt < 4; ++mt)
#pragma unroll
    for (int nt = 0; nt < 2; ++nt)
      acc[mt][nt] = (v8f){0.f, 0.f, 0.f, 0.f, 0.f, 0.f, 0.f, 0.f};

  // global->LDS mapping: each thread moves 32B per matrix per K-step
  const int lr = tid >> 1;                          // tile row / col index 0..127
  const int lc = (tid & 1) * 16;                    // 0 or 16 within BK
  const unsigned short* gA = Ag  + (size_t)lr * p.K + lc;
  const unsigned short* gB = Btg + (size_t)lr * p.K + lc;
  const int lof = lr * LDT + lc;                    // LDS element offset of this chunk

  // fragment addressing (contiguous k: lane half h holds k 16h..16h+15)
  const int arow = wm * 64 + (lane & 15);
  const int kA   = (lane < 16) ? 0 : 16;
  const int bcol = wn * 32 + (lane & 15);
  const int kB   = (lane < 16) ? 0 : 16;

  union Frag { uint4 u[2]; v16bf v; };

#if USE_ASYNC_LDS
  // ---- CDNA5 async DMA path: ASYNCcnt-tracked global->LDS, no VGPR staging
  {
    __builtin_amdgcn_global_load_async_to_lds_b128(GPTR(gA),     LPTR(&sA[0][lof]),     0, 0);
    __builtin_amdgcn_global_load_async_to_lds_b128(GPTR(gA + 8), LPTR(&sA[0][lof + 8]), 0, 0);
    __builtin_amdgcn_global_load_async_to_lds_b128(GPTR(gB),     LPTR(&sB[0][lof]),     0, 0);
    __builtin_amdgcn_global_load_async_to_lds_b128(GPTR(gB + 8), LPTR(&sB[0][lof + 8]), 0, 0);
    __builtin_amdgcn_s_wait_asynccnt(0);
    __syncthreads();
  }
#else
  // ---- fallback: register-staged prologue
  {
    const uint4* ga = (const uint4*)gA;
    const uint4* gb = (const uint4*)gB;
    uint4 a0 = ga[0], a1 = ga[1], b0 = gb[0], b1 = gb[1];
    *(uint4*)(&sA[0][lof])     = a0;
    *(uint4*)(&sA[0][lof + 8]) = a1;
    *(uint4*)(&sB[0][lof])     = b0;
    *(uint4*)(&sB[0][lof + 8]) = b1;
    __syncthreads();
  }
#endif

  for (int k0 = 0; k0 < p.K; k0 += BK) {
    const int cur = (k0 >> 5) & 1;
    const int nxt = cur ^ 1;
    const bool more = (k0 + BK) < p.K;

#if USE_ASYNC_LDS
    if (more) {
      __builtin_amdgcn_global_load_async_to_lds_b128(GPTR(gA + k0 + BK),     LPTR(&sA[nxt][lof]),     0, 0);
      __builtin_amdgcn_global_load_async_to_lds_b128(GPTR(gA + k0 + BK + 8), LPTR(&sA[nxt][lof + 8]), 0, 0);
      __builtin_amdgcn_global_load_async_to_lds_b128(GPTR(gB + k0 + BK),     LPTR(&sB[nxt][lof]),     0, 0);
      __builtin_amdgcn_global_load_async_to_lds_b128(GPTR(gB + k0 + BK + 8), LPTR(&sB[nxt][lof + 8]), 0, 0);
    }
#else
    uint4 a0, a1, b0, b1;
    if (more) {
      const uint4* ga = (const uint4*)(gA + k0 + BK);
      const uint4* gb = (const uint4*)(gB + k0 + BK);
      a0 = ga[0]; a1 = ga[1]; b0 = gb[0]; b1 = gb[1];
    }
#endif

    // ---- compute from current buffer
    const unsigned short* bufA = &sA[cur][0];
    const unsigned short* bufB = &sB[cur][0];
    Frag fa[4];
#pragma unroll
    for (int mt = 0; mt < 4; ++mt) {
      const unsigned short* pa = bufA + (arow + mt * 16) * LDT;
      fa[mt].u[0] = *(const uint4*)(pa + kA);       // K kA..kA+7
      fa[mt].u[1] = *(const uint4*)(pa + 8 + kA);   // K kA+8..kA+15
    }
    Frag fb[2];
#pragma unroll
    for (int nt = 0; nt < 2; ++nt) {
      const unsigned short* pb = bufB + (bcol + nt * 16) * LDT + kB;
      fb[nt].u[0] = *(const uint4*)(pb);            // K kB..kB+7
      fb[nt].u[1] = *(const uint4*)(pb + 8);        // K kB+8..kB+15
    }
#pragma unroll
    for (int mt = 0; mt < 4; ++mt)
#pragma unroll
      for (int nt = 0; nt < 2; ++nt)
        acc[mt][nt] = __builtin_amdgcn_wmma_f32_16x16x32_bf16(
            false, fa[mt].v, false, fb[nt].v, (short)0, acc[mt][nt], false, false);

#if USE_ASYNC_LDS
    __builtin_amdgcn_s_wait_asynccnt(0);
    __syncthreads();
#else
    if (more) {
      *(uint4*)(&sA[nxt][lof])     = a0;
      *(uint4*)(&sA[nxt][lof + 8]) = a1;
      *(uint4*)(&sB[nxt][lof])     = b0;
      *(uint4*)(&sB[nxt][lof + 8]) = b1;
    }
    __syncthreads();
#endif
  }

  // epilogue (C/D layout: VGPR r -> row r, +8 for lanes>=16; col = lane&15)
  const size_t obase = (size_t)bz * (size_t)p.sO;
  const size_t rbase = (size_t)bz * (size_t)p.sR;
  const int m0 = blockIdx.y * BM + wm * 64;
  const int n0 = blockIdx.x * BN + wn * 32;
  const int cn   = lane & 15;
  const int rAdd = (lane >> 4) << 3;
#pragma unroll
  for (int mt = 0; mt < 4; ++mt) {
#pragma unroll
    for (int nt = 0; nt < 2; ++nt) {
#pragma unroll
      for (int r = 0; r < 8; ++r) {
        const int m = m0 + mt * 16 + rAdd + r;
        const int n = n0 + nt * 16 + cn;
        const float val = acc[mt][nt][r];
        if (EPI == 0) {
          p.outBf[obase + (size_t)m * p.Nout + n] = f2bf(val);
        } else if (EPI == 1) {
          const int vb = m >> 11, mm = m & (N_SEQ - 1);   // batch / token
          p.outBf[(size_t)vb * H_DIM * N_SEQ + (size_t)n * N_SEQ + mm] = f2bf(val);
        } else if (EPI == 2) {
          p.outBf[obase + (size_t)m * p.Nout + n] = f2bf(val * p.scale);
        } else if (EPI == 3) {
          float t = val + p.bias[n];
          t = 0.5f * t * (1.0f + erff(t * 0.70710678118654752440f));
          p.outBf[obase + (size_t)m * p.Nout + n] = f2bf(t);
        } else if (EPI == 4) {
          const size_t oi = obase + (size_t)m * p.Nout + n;
          p.outF[oi] = p.resid[oi] + val + p.bias[n];
        } else { // EPI == 5
          const size_t oi = obase + (size_t)m * p.Nout + n;
          const size_t ri = rbase + (size_t)m * p.Nout + n;
          p.outF[oi] = p.resid[ri] + val;
        }
      }
    }
  }
}

// ---------------------------------------------------------------------------
extern "C" void kernel_launch(void* const* d_in, const int* in_sizes, int n_in,
                              void* d_out, int out_size, void* d_ws, size_t ws_size,
                              hipStream_t stream) {
  (void)in_sizes; (void)n_in; (void)out_size; (void)ws_size;
  const float* x    = (const float*)d_in[0];
  const float* ln1g = (const float*)d_in[1];
  const float* ln1b = (const float*)d_in[2];
  const float* Wq   = (const float*)d_in[3];
  const float* Wk   = (const float*)d_in[4];
  const float* Wv   = (const float*)d_in[5];
  const float* ln2g = (const float*)d_in[6];
  const float* ln2b = (const float*)d_in[7];
  const float* W1   = (const float*)d_in[8];
  const float* b1   = (const float*)d_in[9];
  const float* W2   = (const float*)d_in[10];
  const float* b2   = (const float*)d_in[11];
  float* out = (float*)d_out;

  char* ws = (char*)d_ws;
  size_t off = 0;
  auto take = [&](size_t bytes) -> void* {
    void* p = ws + off;
    off += (bytes + 255) & ~(size_t)255;
    return p;
  };
  unsigned short* WqT = (unsigned short*)take((size_t)H_DIM * H_DIM * 2);
  unsigned short* WkT = (unsigned short*)take((size_t)H_DIM * H_DIM * 2);
  unsigned short* WvT = (unsigned short*)take((size_t)H_DIM * H_DIM * 2);
  unsigned short* W1T = (unsigned short*)take((size_t)H_DIM * F_DIM * 2);
  unsigned short* W2T = (unsigned short*)take((size_t)H_DIM * F_DIM * 2);
  unsigned short* yBf = (unsigned short*)take((size_t)T_TOK * H_DIM * 2);
  unsigned short* qBf = (unsigned short*)take((size_t)T_TOK * H_DIM * 2);
  unsigned short* kBf = (unsigned short*)take((size_t)T_TOK * H_DIM * 2);
  unsigned short* vT  = (unsigned short*)take((size_t)T_TOK * H_DIM * 2);
  unsigned short* Sbf = (unsigned short*)take((size_t)B_SZ * N_SEQ * N_SEQ * 2);
  float*          x2  = (float*)take((size_t)T_TOK * H_DIM * 4);
  unsigned short* zBf = (unsigned short*)take((size_t)T_TOK * H_DIM * 2);
  unsigned short* hBf = (unsigned short*)take((size_t)T_TOK * F_DIM * 2);

  const dim3 tb(32, 8);
  transpose_cvt_bf16<<<dim3(H_DIM / 32, H_DIM / 32), tb, 0, stream>>>(Wq, WqT, H_DIM, H_DIM);
  transpose_cvt_bf16<<<dim3(H_DIM / 32, H_DIM / 32), tb, 0, stream>>>(Wk, WkT, H_DIM, H_DIM);
  transpose_cvt_bf16<<<dim3(H_DIM / 32, H_DIM / 32), tb, 0, stream>>>(Wv, WvT, H_DIM, H_DIM);
  transpose_cvt_bf16<<<dim3(F_DIM / 32, H_DIM / 32), tb, 0, stream>>>(W1, W1T, H_DIM, F_DIM);
  transpose_cvt_bf16<<<dim3(H_DIM / 32, F_DIM / 32), tb, 0, stream>>>(W2, W2T, F_DIM, H_DIM);

  layernorm_bf16<<<T_TOK, 256, 0, stream>>>(x, ln1g, ln1b, yBf);

  // q = LN1(x) @ Wq ; k = LN1(x) @ Wk
  GemmParams g{};
  g.A = yBf; g.Bt = WqT; g.outBf = qBf; g.K = H_DIM; g.Nout = H_DIM;
  gemm_bf16_wmma<0><<<dim3(H_DIM / 128, T_TOK / 128, 1), 256, 0, stream>>>(g);
  g.Bt = WkT; g.outBf = kBf;
  gemm_bf16_wmma<0><<<dim3(H_DIM / 128, T_TOK / 128, 1), 256, 0, stream>>>(g);
  // v stored transposed per batch: vT[b][d][token]
  g.Bt = WvT; g.outBf = vT;
  gemm_bf16_wmma<1><<<dim3(H_DIM / 128, T_TOK / 128, 1), 256, 0, stream>>>(g);

  // scores S[b] = q[b] @ k[b]^T * (1/sqrt(H))   (k rows already column-contiguous Bt)
  GemmParams gs{};
  gs.A = qBf; gs.Bt = kBf; gs.outBf = Sbf; gs.K = H_DIM; gs.Nout = N_SEQ;
  gs.sA = (long long)N_SEQ * H_DIM; gs.sB = (long long)N_SEQ * H_DIM;
  gs.sO = (long long)N_SEQ * N_SEQ; gs.scale = 0.03125f;
  gemm_bf16_wmma<2><<<dim3(N_SEQ / 128, N_SEQ / 128, B_SZ), 256, 0, stream>>>(gs);

  softmax_inplace_bf16<<<B_SZ * N_SEQ, 256, 0, stream>>>(Sbf);

  // x2 = x + P @ v
  GemmParams gp{};
  gp.A = Sbf; gp.Bt = vT; gp.outF = x2; gp.resid = x; gp.K = N_SEQ; gp.Nout = H_DIM;
  gp.sA = (long long)N_SEQ * N_SEQ; gp.sB = (long long)H_DIM * N_SEQ;
  gp.sO = (long long)N_SEQ * H_DIM; gp.sR = (long long)N_SEQ * H_DIM;
  gemm_bf16_wmma<5><<<dim3(H_DIM / 128, N_SEQ / 128, B_SZ), 256, 0, stream>>>(gp);

  layernorm_bf16<<<T_TOK, 256, 0, stream>>>(x2, ln2g, ln2b, zBf);

  // h = gelu(z @ W1 + b1)
  GemmParams g1{};
  g1.A = zBf; g1.Bt = W1T; g1.outBf = hBf; g1.bias = b1; g1.K = H_DIM; g1.Nout = F_DIM;
  gemm_bf16_wmma<3><<<dim3(F_DIM / 128, T_TOK / 128, 1), 256, 0, stream>>>(g1);

  // out = x2 + h @ W2 + b2
  GemmParams g2{};
  g2.A = hBf; g2.Bt = W2T; g2.outF = out; g2.bias = b2; g2.resid = x2;
  g2.K = F_DIM; g2.Nout = H_DIM;
  gemm_bf16_wmma<4><<<dim3(H_DIM / 128, T_TOK / 128, 1), 256, 0, stream>>>(g2);
}
